// DecoderLayer_87763361726790
// MI455X (gfx1250) — compile-verified, matrix operand order checked by reading, not run on hardware
//
#include <hip/hip_runtime.h>

// ---------------------------------------------------------------------------
// Decoder layer for MI455X (gfx1250): all matmuls via v_wmma_f32_16x16x32_bf16
// B=2, S=2048, D=1024, H=16, HD=64, DFF=4096.  Wave32, WMMA 16x16x32 bf16.
// GEMM: 128x128 block, 8 waves x (32x64), double-buffered LDS, 1 barrier/iter.
// ---------------------------------------------------------------------------

typedef __attribute__((ext_vector_type(16))) __bf16 v16bf;
typedef __attribute__((ext_vector_type(8)))  __bf16 bf16x8;
typedef __attribute__((ext_vector_type(4)))  __bf16 bf16x4;
typedef __attribute__((ext_vector_type(8)))  float  v8f;
typedef __bf16 bf16;

#define WMMA_BF16(a, b, c) \
  __builtin_amdgcn_wmma_f32_16x16x32_bf16(false, (a), false, (b), (short)0, (c), false, false)

static __device__ inline v8f vzero8() {
  v8f v;
#pragma unroll
  for (int i = 0; i < 8; ++i) v[i] = 0.0f;
  return v;
}

// Load a 16x32 bf16 fragment (A-operand layout; B-operand uses the same
// pattern with its stored N-major row as "row").  Per ISA 7.12.2:
//   lane l (l<16): row = row0 + l, elems 0..7 = K 0..7,  8..15 = K 16..23
//   lane l+16    : same row,       elems 0..7 = K 8..15, 8..15 = K 24..31
static __device__ inline v16bf frag_ld(const bf16* p, int row0, int ldm, int col0, int lane) {
  const bf16* rp = p + (size_t)(row0 + (lane & 15)) * ldm + col0 + ((lane >> 4) << 3);
  bf16x8 lo = *(const bf16x8*)(rp);
  bf16x8 hi = *(const bf16x8*)(rp + 16);
  v16bf f;
#pragma unroll
  for (int i = 0; i < 8; ++i) { f[i] = lo[i]; f[i + 8] = hi[i]; }
  return f;
}

// ---------------------------------------------------------------------------
// f32 -> bf16 conversion. n must be a multiple of 1024 (true for all tensors).
// ---------------------------------------------------------------------------
__global__ void cvt_f32_bf16(const float* __restrict__ in, bf16* __restrict__ out) {
  size_t i = ((size_t)blockIdx.x * 256 + threadIdx.x) * 4;
  float4 v = *(const float4*)(in + i);
  bf16x4 o;
  o[0] = (__bf16)v.x; o[1] = (__bf16)v.y; o[2] = (__bf16)v.z; o[3] = (__bf16)v.w;
  *(bf16x4*)(out + i) = o;
}

// ---------------------------------------------------------------------------
// Tiled WMMA GEMM: C[M,N] = A[M,K] @ W[N,K]^T + bias[N]
// Block tile 128x128, 8 waves (4x2), each wave 32x64 = 2x4 WMMA tiles.
// Double-buffered LDS: global loads for tile k+1 are issued into registers
// while tile k is consumed, committed to the alternate buffer, then ONE
// barrier per iteration.  OUT_MODE: 0 = f32, 1 = bf16, 2 = relu+bf16.
// M % 128 == 0, N % 128 == 0, K % 32 == 0 (all true for this layer).
// ---------------------------------------------------------------------------
#define BM 128
#define BN 128
#define BK 32
#define LDT 40  // padded leading dim in bf16 elems (80B, 16B-aligned)

template <int OUT_MODE>
__launch_bounds__(256)
__global__ void gemm_wmma(const bf16* __restrict__ A, const bf16* __restrict__ W,
                          const float* __restrict__ bias, void* __restrict__ outp,
                          int M, int N, int K) {
  __shared__ __attribute__((aligned(16))) bf16 As[2][BM * LDT];
  __shared__ __attribute__((aligned(16))) bf16 Bs[2][BN * LDT];

  const int tid  = threadIdx.x;
  const int lane = tid & 31;
  const int wid  = tid >> 5;
  const int wrow = wid >> 1;   // 0..3 -> 32-row slices of the 128-row block
  const int wcol = wid & 1;    // 0..1 -> 64-col slices of the 128-col block
  const int mblk = blockIdx.y * BM;
  const int nblk = blockIdx.x * BN;

  // Per-thread staging coordinates: 512 16B vectors per 128x32 tile, 2/thread.
  int sr[2], sc[2];
#pragma unroll
  for (int t = 0; t < 2; ++t) {
    int idx = tid + t * 256;
    sr[t] = idx >> 2;
    sc[t] = (idx & 3) * 8;
  }

  v8f acc[2][4];
#pragma unroll
  for (int m = 0; m < 2; ++m)
#pragma unroll
    for (int n = 0; n < 4; ++n) acc[m][n] = vzero8();

  // Preload tile 0 into buffer 0.
#pragma unroll
  for (int t = 0; t < 2; ++t) {
    *(bf16x8*)(&As[0][sr[t] * LDT + sc[t]]) =
        *(const bf16x8*)(A + (size_t)(mblk + sr[t]) * K + sc[t]);
    *(bf16x8*)(&Bs[0][sr[t] * LDT + sc[t]]) =
        *(const bf16x8*)(W + (size_t)(nblk + sr[t]) * K + sc[t]);
  }
  __syncthreads();

  const int nk = K / BK;
  for (int it = 0; it < nk; ++it) {
    const int cur = it & 1;
    const int nxt = cur ^ 1;
    const bool has_next = (it + 1) < nk;

    // Issue global loads for the next tile (latency hidden under WMMA).
    bf16x8 ra[2], rb[2];
    if (has_next) {
      const int k0 = (it + 1) * BK;
#pragma unroll
      for (int t = 0; t < 2; ++t) {
        ra[t] = *(const bf16x8*)(A + (size_t)(mblk + sr[t]) * K + k0 + sc[t]);
        rb[t] = *(const bf16x8*)(W + (size_t)(nblk + sr[t]) * K + k0 + sc[t]);
      }
    }

    // Consume current tile: 2 A-fragments, 4 B-fragments, 8 WMMAs.
    v16bf af0 = frag_ld(&As[cur][0], wrow * 32,      LDT, 0, lane);
    v16bf af1 = frag_ld(&As[cur][0], wrow * 32 + 16, LDT, 0, lane);
    v16bf bf0 = frag_ld(&Bs[cur][0], wcol * 64,      LDT, 0, lane);
    v16bf bf1 = frag_ld(&Bs[cur][0], wcol * 64 + 16, LDT, 0, lane);
    v16bf bf2 = frag_ld(&Bs[cur][0], wcol * 64 + 32, LDT, 0, lane);
    v16bf bf3 = frag_ld(&Bs[cur][0], wcol * 64 + 48, LDT, 0, lane);

    acc[0][0] = WMMA_BF16(af0, bf0, acc[0][0]);
    acc[0][1] = WMMA_BF16(af0, bf1, acc[0][1]);
    acc[0][2] = WMMA_BF16(af0, bf2, acc[0][2]);
    acc[0][3] = WMMA_BF16(af0, bf3, acc[0][3]);
    acc[1][0] = WMMA_BF16(af1, bf0, acc[1][0]);
    acc[1][1] = WMMA_BF16(af1, bf1, acc[1][1]);
    acc[1][2] = WMMA_BF16(af1, bf2, acc[1][2]);
    acc[1][3] = WMMA_BF16(af1, bf3, acc[1][3]);

    // Commit next tile into the alternate buffer.
    if (has_next) {
#pragma unroll
      for (int t = 0; t < 2; ++t) {
        *(bf16x8*)(&As[nxt][sr[t] * LDT + sc[t]]) = ra[t];
        *(bf16x8*)(&Bs[nxt][sr[t] * LDT + sc[t]]) = rb[t];
      }
    }
    __syncthreads();
  }

  // Epilogue. C layout: col = lane%16, rows r + 8*(lane/16) per vgpr r.
#pragma unroll
  for (int m = 0; m < 2; ++m)
#pragma unroll
    for (int n = 0; n < 4; ++n) {
      int col = nblk + wcol * 64 + n * 16 + (lane & 15);
      float bv = bias[col];
#pragma unroll
      for (int r = 0; r < 8; ++r) {
        int row = mblk + wrow * 32 + m * 16 + ((lane >> 4) << 3) + r;
        float v = acc[m][n][r] + bv;
        if (OUT_MODE == 2) v = fmaxf(v, 0.0f);
        if (OUT_MODE == 0) ((float*)outp)[(size_t)row * N + col] = v;
        else               ((bf16*)outp)[(size_t)row * N + col] = (__bf16)v;
      }
    }
}

// ---------------------------------------------------------------------------
// Flash attention (online softmax), bf16 WMMA.
// qkv: (B*S, 3*D) bf16 where feature = comp*D + h*64 + hd.
// Grid (S/64, H, B); block 128 = 4 waves; wave owns a 16-query tile.
// causal=1: arithmetic tril mask. causal=0: per-batch key mask (enc_out).
// ---------------------------------------------------------------------------
#define ATT_S  2048
#define ATT_D  1024
#define ATT_D3 3072

__launch_bounds__(128)
__global__ void attn_wmma(const bf16* __restrict__ qkv, const int* __restrict__ kmask,
                          bf16* __restrict__ out, int causal) {
  __shared__ __attribute__((aligned(16))) bf16 Ks[32 * 72];      // 32 keys x 64 d
  __shared__ __attribute__((aligned(16))) bf16 Vt[64 * LDT];     // transposed: 64 d x 32 keys
  __shared__ __attribute__((aligned(16))) bf16 Ps[4][16 * LDT];  // per-wave P tile

  const int tid  = threadIdx.x;
  const int lane = tid & 31;
  const int wid  = tid >> 5;
  const int qb = blockIdx.x, h = blockIdx.y, b = blockIdx.z;
  const int q0 = qb * 64 + wid * 16;

  const size_t base = (size_t)b * ATT_S * ATT_D3 + (size_t)h * 64;
  const bf16* Qb = qkv + base;                 // comp 0
  const bf16* Kb = qkv + base + ATT_D;         // comp 1
  const bf16* Vb = qkv + base + 2 * ATT_D;     // comp 2

  v16bf qf0 = frag_ld(Qb, q0, ATT_D3, 0,  lane);
  v16bf qf1 = frag_ld(Qb, q0, ATT_D3, 32, lane);

  v8f o0 = vzero8(), o1 = vzero8(), o2 = vzero8(), o3 = vzero8();
  float rm[8], rs[8];
#pragma unroll
  for (int r = 0; r < 8; ++r) { rm[r] = -1e30f; rs[r] = 0.0f; }

  const int nkb = causal ? (qb + 1) * 2 : (ATT_S / 32);

  for (int kb = 0; kb < nkb; ++kb) {
    const int kst = kb * 32;
    // Stage K tile (32 keys x 64 d), row-major (key = B-operand N row).
#pragma unroll
    for (int t = 0; t < 2; ++t) {
      int idx = tid + t * 128;
      int r = idx >> 3, c8 = (idx & 7) * 8;
      *(bf16x8*)(Ks + r * 72 + c8) =
          *(const bf16x8*)(Kb + (size_t)(kst + r) * ATT_D3 + c8);
    }
    // Stage V transposed (d x key) so d is the N-major row for the PV matmul.
#pragma unroll
    for (int t = 0; t < 2; ++t) {
      int idx = tid + t * 128;
      int r = idx >> 3, c8 = (idx & 7) * 8;
      bf16x8 v = *(const bf16x8*)(Vb + (size_t)(kst + r) * ATT_D3 + c8);
#pragma unroll
      for (int j = 0; j < 8; ++j) Vt[(c8 + j) * LDT + r] = v[j];
    }
    // Prefetch next key block into cache (global_prefetch_b8); the long
    // softmax VALU stretch below hides the latency.
    if (kb + 1 < nkb) {
      int idx = tid;
      int r = idx >> 3, c8 = (idx & 7) * 8;
      __builtin_prefetch(Kb + (size_t)(kst + 32 + r) * ATT_D3 + c8, 0, 0);
      __builtin_prefetch(Vb + (size_t)(kst + 32 + r) * ATT_D3 + c8, 0, 0);
    }
    __syncthreads();

    // S = Q @ K^T over HD=64 (two 32-deep wmma steps), two 16-key subtiles.
    v8f s0 = vzero8(), s1 = vzero8();
    {
      v16bf k00 = frag_ld(Ks, 0,  72, 0,  lane);
      v16bf k01 = frag_ld(Ks, 0,  72, 32, lane);
      v16bf k10 = frag_ld(Ks, 16, 72, 0,  lane);
      v16bf k11 = frag_ld(Ks, 16, 72, 32, lane);
      s0 = WMMA_BF16(qf0, k00, s0);
      s0 = WMMA_BF16(qf1, k01, s0);
      s1 = WMMA_BF16(qf0, k10, s1);
      s1 = WMMA_BF16(qf1, k11, s1);
    }

    const int c0 = kst + (lane & 15);
    const int c1 = c0 + 16;
    float mk0 = 0.0f, mk1 = 0.0f;
    if (!causal) {
      mk0 = (kmask[(size_t)b * ATT_S + c0] != 0) ? 0.0f : -1e30f;
      mk1 = (kmask[(size_t)b * ATT_S + c1] != 0) ? 0.0f : -1e30f;
    }

    float p0[8], p1[8], mloc[8];
#pragma unroll
    for (int r = 0; r < 8; ++r) {
      int qr = q0 + r + ((lane >> 4) << 3);
      float v0 = s0[r] * 0.125f;  // 1/sqrt(64)
      float v1 = s1[r] * 0.125f;
      if (causal) {
        if (c0 > qr) v0 = -1e30f;
        if (c1 > qr) v1 = -1e30f;
      } else {
        v0 += mk0; v1 += mk1;
      }
      p0[r] = v0; p1[r] = v1;
      float m = fmaxf(v0, v1);   // row-wise max across the 16-lane half
      m = fmaxf(m, __shfl_xor(m, 1, 32));
      m = fmaxf(m, __shfl_xor(m, 2, 32));
      m = fmaxf(m, __shfl_xor(m, 4, 32));
      m = fmaxf(m, __shfl_xor(m, 8, 32));
      mloc[r] = m;
    }

#pragma unroll
    for (int r = 0; r < 8; ++r) {
      float mnew = fmaxf(rm[r], mloc[r]);
      float corr = __expf(rm[r] - mnew);
      float e0 = __expf(p0[r] - mnew);
      float e1 = __expf(p1[r] - mnew);
      float ls = e0 + e1;
      ls += __shfl_xor(ls, 1, 32);
      ls += __shfl_xor(ls, 2, 32);
      ls += __shfl_xor(ls, 4, 32);
      ls += __shfl_xor(ls, 8, 32);
      rs[r] = rs[r] * corr + ls;
      rm[r] = mnew;
      o0[r] *= corr; o1[r] *= corr; o2[r] *= corr; o3[r] *= corr;
      int prow = r + ((lane >> 4) << 3);
      Ps[wid][prow * LDT + (lane & 15)]      = (__bf16)e0;
      Ps[wid][prow * LDT + 16 + (lane & 15)] = (__bf16)e1;
    }

    // O += P(16x32) @ V(32x64): 4 N-subtiles of 16 d each.
    v16bf pf  = frag_ld(Ps[wid], 0,  LDT, 0, lane);
    v16bf v0f = frag_ld(Vt, 0,  LDT, 0, lane);
    v16bf v1f = frag_ld(Vt, 16, LDT, 0, lane);
    v16bf v2f = frag_ld(Vt, 32, LDT, 0, lane);
    v16bf v3f = frag_ld(Vt, 48, LDT, 0, lane);
    o0 = WMMA_BF16(pf, v0f, o0);
    o1 = WMMA_BF16(pf, v1f, o1);
    o2 = WMMA_BF16(pf, v2f, o2);
    o3 = WMMA_BF16(pf, v3f, o3);
    __syncthreads();
  }

  // Normalize + store bf16 (feature = h*64 + d).
#pragma unroll
  for (int r = 0; r < 8; ++r) {
    int qr = q0 + r + ((lane >> 4) << 3);
    size_t off = ((size_t)b * ATT_S + qr) * ATT_D + (size_t)h * 64 + (lane & 15);
    float inv = 1.0f / rs[r];
    out[off + 0]  = (__bf16)(o0[r] * inv);
    out[off + 16] = (__bf16)(o1[r] * inv);
    out[off + 32] = (__bf16)(o2[r] * inv);
    out[off + 48] = (__bf16)(o3[r] * inv);
  }
}

// ---------------------------------------------------------------------------
// Fused residual add + LayerNorm over D=1024. One row per block, 256 threads.
// Writes f32 (residual chain / final output) and optional bf16 (GEMM feed).
// ---------------------------------------------------------------------------
__launch_bounds__(256)
__global__ void add_ln(const float* __restrict__ a, const float* __restrict__ res,
                       const float* __restrict__ g, const float* __restrict__ beta,
                       float* __restrict__ of, bf16* __restrict__ ob) {
  const int Dn = 1024;
  const size_t row = blockIdx.x;
  const int c = threadIdx.x * 4;
  const int lane = threadIdx.x & 31, wid = threadIdx.x >> 5;

  float4 va = *(const float4*)(a + row * Dn + c);
  float4 vr = *(const float4*)(res + row * Dn + c);
  float v0 = va.x + vr.x, v1 = va.y + vr.y, v2 = va.z + vr.z, v3 = va.w + vr.w;
  float sum = v0 + v1 + v2 + v3;
  float sq  = v0 * v0 + v1 * v1 + v2 * v2 + v3 * v3;
#pragma unroll
  for (int m = 1; m < 32; m <<= 1) {
    sum += __shfl_xor(sum, m, 32);
    sq  += __shfl_xor(sq,  m, 32);
  }
  __shared__ float ssum[8], ssq[8];
  if (lane == 0) { ssum[wid] = sum; ssq[wid] = sq; }
  __syncthreads();
  float tot = 0.0f, totq = 0.0f;
#pragma unroll
  for (int i = 0; i < 8; ++i) { tot += ssum[i]; totq += ssq[i]; }

  const float mean = tot * (1.0f / Dn);
  const float var  = totq * (1.0f / Dn) - mean * mean;
  const float inv  = rsqrtf(var + 1e-5f);

  float4 vg = *(const float4*)(g + c);
  float4 vb = *(const float4*)(beta + c);
  float y0 = (v0 - mean) * inv * vg.x + vb.x;
  float y1 = (v1 - mean) * inv * vg.y + vb.y;
  float y2 = (v2 - mean) * inv * vg.z + vb.z;
  float y3 = (v3 - mean) * inv * vg.w + vb.w;

  float4 o4; o4.x = y0; o4.y = y1; o4.z = y2; o4.w = y3;
  *(float4*)(of + row * Dn + c) = o4;
  if (ob) {
    bf16x4 ob4;
    ob4[0] = (__bf16)y0; ob4[1] = (__bf16)y1; ob4[2] = (__bf16)y2; ob4[3] = (__bf16)y3;
    *(bf16x4*)(ob + row * Dn + c) = ob4;
  }
}

// ---------------------------------------------------------------------------
// Host-side orchestration (all launches on `stream`, graph-capture safe).
// ---------------------------------------------------------------------------
extern "C" void kernel_launch(void* const* d_in, const int* in_sizes, int n_in,
                              void* d_out, int out_size, void* d_ws, size_t ws_size,
                              hipStream_t stream) {
  (void)in_sizes; (void)n_in; (void)out_size; (void)ws_size;
  constexpr int Bc = 2, Sc = 2048, Dc = 1024, DFFc = 4096;
  constexpr int Mr = Bc * Sc;        // 4096 rows
  constexpr int TD = 3 * Dc;         // 3072

  const float* x       = (const float*)d_in[0];
  const int*   enc_out = (const int*)d_in[1];
  // d_in[2] src_mask (unused by reference path), d_in[3] trg_mask == tril (arithmetic causal)
  const float* qkv_w1 = (const float*)d_in[4];
  const float* qkv_b1 = (const float*)d_in[5];
  const float* out_w1 = (const float*)d_in[6];
  const float* out_b1 = (const float*)d_in[7];
  const float* qkv_w2 = (const float*)d_in[8];
  const float* qkv_b2 = (const float*)d_in[9];
  const float* out_w2 = (const float*)d_in[10];
  const float* out_b2 = (const float*)d_in[11];
  const float* ln1_g  = (const float*)d_in[12];
  const float* ln1_b  = (const float*)d_in[13];
  const float* ln2_g  = (const float*)d_in[14];
  const float* ln2_b  = (const float*)d_in[15];
  const float* ln3_g  = (const float*)d_in[16];
  const float* ln3_b  = (const float*)d_in[17];
  const float* ff_w1  = (const float*)d_in[18];
  const float* ff_b1  = (const float*)d_in[19];
  const float* ff_w2  = (const float*)d_in[20];
  const float* ff_b2  = (const float*)d_in[21];

  // Workspace carve-up (bump allocator, 256B aligned).
  char* cur = (char*)d_ws;
  auto alloc = [&](size_t bytes) -> void* {
    void* p = (void*)cur;
    cur += (bytes + 255) & ~(size_t)255;
    return p;
  };
  bf16* wqkv1 = (bf16*)alloc((size_t)TD * Dc * 2);
  bf16* wout1 = (bf16*)alloc((size_t)Dc * Dc * 2);
  bf16* wqkv2 = (bf16*)alloc((size_t)TD * Dc * 2);
  bf16* wout2 = (bf16*)alloc((size_t)Dc * Dc * 2);
  bf16* wff1  = (bf16*)alloc((size_t)DFFc * Dc * 2);
  bf16* wff2  = (bf16*)alloc((size_t)Dc * DFFc * 2);
  bf16* x_bf   = (bf16*)alloc((size_t)Mr * Dc * 2);
  bf16* qkv_a  = (bf16*)alloc((size_t)Mr * TD * 2);
  bf16* attn_a = (bf16*)alloc((size_t)Mr * Dc * 2);
  float* tmp_f = (float*)alloc((size_t)Mr * Dc * 4);
  float* x1_f  = (float*)alloc((size_t)Mr * Dc * 4);
  bf16*  x1_bf = (bf16*)alloc((size_t)Mr * Dc * 2);
  float* x2_f  = (float*)alloc((size_t)Mr * Dc * 4);
  bf16*  x2_bf = (bf16*)alloc((size_t)Mr * Dc * 2);
  bf16*  h1_bf = (bf16*)alloc((size_t)Mr * DFFc * 2);

  auto cvt = [&](const float* src, bf16* dst, size_t n) {
    cvt_f32_bf16<<<dim3((unsigned)(n / 1024)), 256, 0, stream>>>(src, dst);
  };
  cvt(x, x_bf, (size_t)Mr * Dc);
  cvt(qkv_w1, wqkv1, (size_t)TD * Dc);
  cvt(out_w1, wout1, (size_t)Dc * Dc);
  cvt(qkv_w2, wqkv2, (size_t)TD * Dc);
  cvt(out_w2, wout2, (size_t)Dc * Dc);
  cvt(ff_w1, wff1, (size_t)DFFc * Dc);
  cvt(ff_w2, wff2, (size_t)Dc * DFFc);

  const dim3 blk256(256), blk128(128);
  auto gemm_grid = [](int M, int N) { return dim3((unsigned)(N / BN), (unsigned)(M / BM)); };
  const dim3 attn_grid(Sc / 64, 16, Bc);
  const dim3 ln_grid(Mr);

  // ---- Self-attention block ----
  gemm_wmma<1><<<gemm_grid(Mr, TD), blk256, 0, stream>>>(x_bf, wqkv1, qkv_b1, qkv_a, Mr, TD, Dc);
  attn_wmma<<<attn_grid, blk128, 0, stream>>>(qkv_a, nullptr, attn_a, /*causal=*/1);
  gemm_wmma<0><<<gemm_grid(Mr, Dc), blk256, 0, stream>>>(attn_a, wout1, out_b1, tmp_f, Mr, Dc, Dc);
  add_ln<<<ln_grid, blk256, 0, stream>>>(tmp_f, x, ln1_g, ln1_b, x1_f, x1_bf);

  // ---- "Cross"-attention block (Q=K=V from x1, key mask = enc_out) ----
  gemm_wmma<1><<<gemm_grid(Mr, TD), blk256, 0, stream>>>(x1_bf, wqkv2, qkv_b2, qkv_a, Mr, TD, Dc);
  attn_wmma<<<attn_grid, blk128, 0, stream>>>(qkv_a, enc_out, attn_a, /*causal=*/0);
  gemm_wmma<0><<<gemm_grid(Mr, Dc), blk256, 0, stream>>>(attn_a, wout2, out_b2, tmp_f, Mr, Dc, Dc);
  add_ln<<<ln_grid, blk256, 0, stream>>>(tmp_f, x1_f, ln2_g, ln2_b, x2_f, x2_bf);

  // ---- FFN ----
  gemm_wmma<2><<<gemm_grid(Mr, DFFc), blk256, 0, stream>>>(x2_bf, wff1, ff_b1, h1_bf, Mr, DFFc, Dc);
  gemm_wmma<0><<<gemm_grid(Mr, Dc), blk256, 0, stream>>>(h1_bf, wff2, ff_b2, tmp_f, Mr, Dc, DFFc);
  add_ln<<<ln_grid, blk256, 0, stream>>>(tmp_f, x2_f, ln3_g, ln3_b, (float*)d_out, nullptr);
}
